// GraphAttentionLayer_42442866819879
// MI455X (gfx1250) — matrix operand order repaired and checked by reading, not compiled, not run on hardware
//
#include <hip/hip_runtime.h>

typedef __attribute__((ext_vector_type(2)))  float    v2f;
typedef __attribute__((ext_vector_type(8)))  float    v8f;
typedef __attribute__((ext_vector_type(8)))  _Float16 v8h;
typedef __attribute__((ext_vector_type(16))) _Float16 v16h;

#define N_NODES 4096
#define IN_F    256
#define OUT_F   128
#define ALPHA   0.2f

// -------------------------------------------------------------------------
// Kernel 1: wh = lstm_out @ W   (fp32, exact, via v_wmma_f32_16x16x4_f32)
// Also emits whT = wh^T in f16 for the attention GEMM (contiguous K reads).
// Block = 256 threads = 8 waves; wave w computes 16x16 tile (rows i0..i0+15,
// cols 16w..16w+15). Grid = 256 blocks (one per 16-row block).
// -------------------------------------------------------------------------
__global__ void k_gemm_wh(const float* __restrict__ X,   // [4096][256]
                          const float* __restrict__ W,   // [256][128]
                          float* __restrict__ wh,        // [4096][128] fp32
                          _Float16* __restrict__ whT)    // [128][4096] f16
{
    const int lane = threadIdx.x & 31;
    const int wave = threadIdx.x >> 5;          // 0..7  -> column block
    const int i0   = blockIdx.x * 16;           // row block
    const int lg   = lane >> 4;                 // 0/1 lane group
    const int m    = lane & 15;
    const int col  = wave * 16 + m;             // N index for B/C/D

    v8f c = {};
    const float* Arow = X + (size_t)(i0 + m) * IN_F;   // A: M = m

    for (int kb = 0; kb < IN_F; kb += 4) {
        const int k = kb + 2 * lg;              // A/B K pair base per ISA layout
        v2f a, b;
        a.x = Arow[k];
        a.y = Arow[k + 1];
        b.x = W[(size_t)k * OUT_F + col];
        b.y = W[(size_t)(k + 1) * OUT_F + col];
        c = __builtin_amdgcn_wmma_f32_16x16x4_f32(
                false, a, false, b, (short)0, c, false, false);
    }

    // C/D layout: VGPR r holds row (r + 8*lg), col = col
#pragma unroll
    for (int r = 0; r < 8; ++r) {
        const int row = i0 + r + 8 * lg;
        wh[(size_t)row * OUT_F + col] = c[r];
    }
    v8h h;
#pragma unroll
    for (int r = 0; r < 8; ++r) h[r] = (_Float16)c[r];
    // rows r..r+7 contiguous in whT[col][*]; 16B aligned (i0 mult of 16)
    *(v8h*)(whT + (size_t)col * N_NODES + i0 + 8 * lg) = h;
}

// -------------------------------------------------------------------------
// Kernel 2: si = wh @ a1, sj = wh @ a3  (one thread per row)
// -------------------------------------------------------------------------
__global__ void k_sisj(const float* __restrict__ wh,
                       const float* __restrict__ a,     // [261]
                       float* __restrict__ si,
                       float* __restrict__ sj)
{
    const int row = blockIdx.x * blockDim.x + threadIdx.x;
    if (row >= N_NODES) return;
    const float* r = wh + (size_t)row * OUT_F;
    float s1 = 0.f, s3 = 0.f;
#pragma unroll 4
    for (int c = 0; c < OUT_F; ++c) {
        const float v = r[c];
        s1 += v * a[c];                 // a1 = a[0:128]
        s3 += v * a[OUT_F + 5 + c];     // a3 = a[133:261]
    }
    si[row] = s1;
    sj[row] = s3;
}

// -------------------------------------------------------------------------
// Kernel 3: fused score -> leaky -> mask -> exp -> (P @ wh) -> /rowsum
// Block handles 16 rows; 8 waves each own 16 output columns.
// Column loop: tiles of 32 (WMMA K). Single pass over edges/adj (HBM-bound).
// exp(score) never overflows fp32 here, so no max-subtraction needed;
// masked entries contribute exactly 0, matching softmax(where(adj, s, -9e15)).
// -------------------------------------------------------------------------
__global__ void k_attn(const float* __restrict__ edges,   // [N*N][5]
                       const int*   __restrict__ adj,     // [N][N]
                       const float* __restrict__ si,
                       const float* __restrict__ sj,
                       const float* __restrict__ a,       // a2 = a[128:133]
                       const _Float16* __restrict__ whT,  // [128][4096] f16
                       float* __restrict__ out)           // [4096][128]
{
    __shared__ _Float16 ldsP[16][32];     // P tile (f16) for WMMA A-fragment
    __shared__ float    ldsRed[16][16];   // per-row partial sums
    __shared__ float    lsum[16];         // row softmax denominators

    const int t    = threadIdx.x;
    const int lane = t & 31;
    const int wave = t >> 5;
    const int lg   = lane >> 4;
    const int m    = lane & 15;
    const int i    = t >> 4;              // 0..15: row this thread scores
    const int jl   = t & 15;              // 0..15: column slot within tile
    const int i0   = blockIdx.x * 16;

    float a2[5];
#pragma unroll
    for (int e = 0; e < 5; ++e) a2[e] = a[OUT_F + e];

    const float  sbase = si[i0 + i];
    const float* Er    = edges + (size_t)(i0 + i) * N_NODES * 5;
    const int*   Ar    = adj   + (size_t)(i0 + i) * N_NODES;

    const int        col  = wave * 16 + m;                 // output column
    const _Float16*  Bcol = whT + (size_t)col * N_NODES;   // B: K contiguous

    float lacc = 0.f;
    v8f   o    = {};

    for (int J = 0; J < N_NODES; J += 32) {
#pragma unroll
        for (int h = 0; h < 2; ++h) {
            const int    j  = J + jl + 16 * h;
            const float* e5 = Er + (size_t)j * 5;
            const float  rr = e5[0]*a2[0] + e5[1]*a2[1] + e5[2]*a2[2]
                            + e5[3]*a2[3] + e5[4]*a2[4];
            float s = sbase + sj[j] + rr;
            s = (s > 0.f) ? s : ALPHA * s;
            const float p = (Ar[j] > 0) ? __expf(s) : 0.f;
            lacc += p;
            ldsP[i][jl + 16 * h] = (_Float16)p;
        }
        __syncthreads();

        // A-fragment (16x32 f16): lane m holds K = 8*lg..+7 and 16+8*lg..+7
        const _Float16* prow = &ldsP[m][8 * lg];
        const v8h alo = *(const v8h*)prow;          // ds_load_b128
        const v8h ahi = *(const v8h*)(prow + 16);   // ds_load_b128
        v16h afrag;
#pragma unroll
        for (int q = 0; q < 8; ++q) { afrag[q] = alo[q]; afrag[8 + q] = ahi[q]; }

        // B-fragment (32x16 f16): lane holds K = 8*lg..+7 and 16+8*lg..+7 at column col
        const _Float16* bp = Bcol + J + 8 * lg;
        const v8h blo = *(const v8h*)bp;            // global_load_b128 (L2-hot)
        const v8h bhi = *(const v8h*)(bp + 16);
        v16h bfrag;
#pragma unroll
        for (int q = 0; q < 8; ++q) { bfrag[q] = blo[q]; bfrag[8 + q] = bhi[q]; }

        o = __builtin_amdgcn_wmma_f32_16x16x32_f16(
                false, afrag, false, bfrag, (short)0, o, false, false);
        __syncthreads();   // protect ldsP before next tile's writes
    }

    // Row-sum reduction (16 partials per row)
    ldsRed[i][jl] = lacc;
    __syncthreads();
    if (t < 16) {
        float s = 0.f;
#pragma unroll
        for (int q = 0; q < 16; ++q) s += ldsRed[t][q];
        lsum[t] = s;
    }
    __syncthreads();

    // Normalize and store: VGPR r holds row (r + 8*lg)
#pragma unroll
    for (int r = 0; r < 8; ++r) {
        const int row = r + 8 * lg;
        out[(size_t)(i0 + row) * OUT_F + col] = o[r] / lsum[row];
    }
}

// -------------------------------------------------------------------------
// Host launcher. Input order: ids, lstm_out, edges_list, adj_mat, first, W, a
// Workspace: wh fp32 (2MB) | whT f16 (1MB) | si (16KB) | sj (16KB)
// -------------------------------------------------------------------------
extern "C" void kernel_launch(void* const* d_in, const int* in_sizes, int n_in,
                              void* d_out, int out_size, void* d_ws, size_t ws_size,
                              hipStream_t stream) {
    (void)in_sizes; (void)n_in; (void)out_size; (void)ws_size;

    const float* lstm  = (const float*)d_in[1];
    const float* edges = (const float*)d_in[2];
    const int*   adj   = (const int*)  d_in[3];
    const float* W     = (const float*)d_in[5];
    const float* a     = (const float*)d_in[6];
    float*       out   = (float*)d_out;

    char*      ws  = (char*)d_ws;
    float*     wh  = (float*)ws;                                   // 2 MB
    _Float16*  whT = (_Float16*)(ws + (size_t)N_NODES * OUT_F * 4); // 1 MB
    float*     si  = (float*)(ws + (size_t)3 * 1024 * 1024);        // 16 KB
    float*     sj  = si + N_NODES;                                  // 16 KB

    k_gemm_wh<<<N_NODES / 16, 256, 0, stream>>>(lstm, W, wh, whT);
    k_sisj  <<<N_NODES / 256, 256, 0, stream>>>(wh, a, si, sj);
    k_attn  <<<N_NODES / 16, 256, 0, stream>>>(edges, adj, si, sj, a, whT, out);
}
